// LfadsCellModifiedGru_53575422050498
// MI455X (gfx1250) — matrix operand order repaired and checked by reading, not compiled, not run on hardware
//
#include <hip/hip_runtime.h>
#include <hip/hip_bf16.h>

typedef __bf16 bf16_t;
typedef __attribute__((ext_vector_type(16))) __bf16 v16bf;
typedef __attribute__((ext_vector_type(8)))  __bf16 v8bf;
typedef __attribute__((ext_vector_type(8)))  float  v8f;

#define B_   256
#define T_   512
#define IN_  128
#define EH_  512
#define GH_  512
#define KENC 640                 // IN_ + EH_ (fused K for encoder step GEMM)
#define LDSE 648                 // KENC + 8 pad (4-bank stagger per row)
#define LDSG 520                 // GH_ + 8 pad
#define NBLK 64                  // 8 M-blocks x 8 J-groups
#define UPDATE_BIAS_ 1.0f
#define CLIP_VAL_    5.0f

// ---- workspace layout (bytes, all 256-aligned) ----
#define WS_WENC  0                          // bf16 [1536][640]  (rows: r|z|n, cols: [Wi|Wh])
#define WS_WRU   (WS_WENC + 1536*640*2)     // bf16 [1024][512]
#define WS_WC    (WS_WRU  + 1024*512*2)     // bf16 [512][512]
#define WS_H0    (WS_WC   + 512*512*2)      // f32 [256][512] enc state (double buffered)
#define WS_H1    (WS_H0   + 256*512*4)
#define WS_G0    (WS_H1   + 256*512*4)      // f32 gen state (double buffered)
#define WS_G1    (WS_G0   + 256*512*4)
#define WS_RH    (WS_G1   + 256*512*4)      // f32 [256][512] r*h intermediate
#define WS_BAR   (WS_RH   + 256*512*4)      // 4 x u32 barrier state

// ---------------- helpers ----------------
__device__ __forceinline__ bf16_t f2bf(float f) {
  unsigned u = __builtin_bit_cast(unsigned, f);
  u += 0x7fffu + ((u >> 16) & 1u);                 // round-to-nearest-even
  unsigned short s = (unsigned short)(u >> 16);
  return __builtin_bit_cast(bf16_t, s);
}

__device__ __forceinline__ v16bf cat8(v8bf lo, v8bf hi) {
  return __builtin_shufflevector(lo, hi, 0,1,2,3,4,5,6,7,8,9,10,11,12,13,14,15);
}

// A-operand 16x32 bf16 fragment (ISA 7.12.2 16-bit A layout):
// lane L: M=L%16, elems 0..7 -> K=[8*(L/16), +8), elems 8..15 -> K=[16+8*(L/16), +8)
__device__ __forceinline__ v16bf load_fragA(const bf16_t* base, int row0, int k0,
                                            int stride, int lane) {
  int m = lane & 15, h = lane >> 4;
  const bf16_t* p = base + (size_t)(row0 + m) * stride + k0 + h * 8;
  v8bf lo = *(const v8bf*)p;
  v8bf hi = *(const v8bf*)(p + 16);
  return cat8(lo, hi);
}

// B-operand 32x16 fragment from weight rows W[N][K] (B = W^T tile):
// lane L: N=L%16, elems 0..7 -> K=k0+[8*(L/16), +8), elems 8..15 -> K=k0+[16+8*(L/16), +8)
__device__ __forceinline__ v16bf load_fragB(const bf16_t* W, int n0, int k0,
                                            int stride, int lane) {
  const bf16_t* p = W + (size_t)(n0 + (lane & 15)) * stride + k0 + (lane >> 4) * 8;
  v8bf lo = *(const v8bf*)p;
  v8bf hi = *(const v8bf*)(p + 16);
  return cat8(lo, hi);
}

__device__ __forceinline__ v8f wmma_bf16(v16bf a, v16bf b, v8f c) {
  return __builtin_amdgcn_wmma_f32_16x16x32_bf16(false, a, false, b, (short)0, c,
                                                 false, false);
}

__device__ __forceinline__ float sigmoidf_(float x) { return 1.0f / (1.0f + __expf(-x)); }

// generation-counter grid barrier (persistent-kernel RNN pattern)
__device__ __forceinline__ void grid_barrier(unsigned* cnt, unsigned* gen, unsigned nb) {
  __threadfence();
  __syncthreads();
  if (threadIdx.x == 0) {
    unsigned g = __hip_atomic_load(gen, __ATOMIC_RELAXED, __HIP_MEMORY_SCOPE_AGENT);
    unsigned a = __hip_atomic_fetch_add(cnt, 1u, __ATOMIC_ACQ_REL, __HIP_MEMORY_SCOPE_AGENT);
    if (a == nb - 1u) {
      __hip_atomic_store(cnt, 0u, __ATOMIC_RELAXED, __HIP_MEMORY_SCOPE_AGENT);
      __hip_atomic_fetch_add(gen, 1u, __ATOMIC_ACQ_REL, __HIP_MEMORY_SCOPE_AGENT);
    } else {
      while (__hip_atomic_load(gen, __ATOMIC_ACQUIRE, __HIP_MEMORY_SCOPE_AGENT) == g)
        __builtin_amdgcn_s_sleep(1);
    }
  }
  __syncthreads();
}

// ---------------- kernels ----------------
__global__ void prep_kernel(const float* __restrict__ enc_Wi, const float* __restrict__ enc_Wh,
                            const float* __restrict__ gWru, const float* __restrict__ gWc,
                            bf16_t* Wenc, bf16_t* Wru, bf16_t* Wc, float* h0, unsigned* bar) {
  int gid = blockIdx.x * blockDim.x + threadIdx.x;
  int str = gridDim.x * blockDim.x;
  if (gid < 4) bar[gid] = 0u;
  for (int i = gid; i < 1536 * KENC; i += str) {
    int n = i / KENC, k = i % KENC;
    float w = (k < IN_) ? enc_Wi[n * IN_ + k] : enc_Wh[n * EH_ + (k - IN_)];
    Wenc[i] = f2bf(w);
  }
  for (int i = gid; i < 1024 * GH_; i += str) Wru[i] = f2bf(gWru[i]);
  for (int i = gid; i < GH_ * GH_; i += str)  Wc[i]  = f2bf(gWc[i]);
  for (int i = gid; i < B_ * EH_; i += str)   h0[i]  = 0.0f;
}

__global__ void __launch_bounds__(256)
enc_kernel(const float* __restrict__ src, const bf16_t* __restrict__ Wenc,
           float* hbuf0, float* hbuf1, unsigned* bar) {
  __shared__ __align__(16) bf16_t ldsW[192 * LDSE];  // [r|z|n] x 64 rows, K=640
  __shared__ __align__(16) bf16_t ldsA[32 * LDSE];   // [x_t | h] block
  const int tid = threadIdx.x;
  const int wgM = blockIdx.x >> 3, wgJ = blockIdx.x & 7;
  const int rowbase = wgM * 32;
  const int w = tid >> 5, lane = tid & 31;
  const int mrow = (w & 1) * 16;        // wave M-tile within block
  const int jloc = (w >> 1) * 16;       // wave N-tile within WG's 64 cols
  const int jglb = wgJ * 64 + jloc;

  // stage WG's weight slice into LDS once (loop-invariant home = LDS, not VGPRs)
  for (int i = tid; i < 192 * (KENC / 4); i += 256) {
    int rr = i / (KENC / 4);
    int k4 = (i - rr * (KENC / 4)) * 4;
    int g = rr >> 6, r64 = rr & 63;
    const bf16_t* gp = Wenc + (size_t)(g * 512 + wgJ * 64 + r64) * KENC + k4;
    *(uint2*)(ldsW + rr * LDSE + k4) = *(const uint2*)gp;
  }
  __syncthreads();

#pragma unroll 1
  for (int t = 0; t < T_; ++t) {
    const float* hcur = (t & 1) ? hbuf1 : hbuf0;
    float*       hnxt = (t & 1) ? hbuf0 : hbuf1;

    // stage A = [x_t | h] as bf16 into LDS
    for (int i = tid; i < 32 * IN_; i += 256) {
      int r = i >> 7, c = i & (IN_ - 1);
      ldsA[r * LDSE + c] =
          f2bf(src[(size_t)(rowbase + r) * (T_ * IN_) + (size_t)t * IN_ + c]);
    }
    for (int i = tid; i < 32 * EH_; i += 256) {
      int r = i >> 9, c = i & (EH_ - 1);
      ldsA[r * LDSE + IN_ + c] = f2bf(hcur[(rowbase + r) * EH_ + c]);
    }
    __syncthreads();

    v8f accR = {}, accZ = {}, accNX = {}, accNH = {};
    // K in [0,128): input projection part, n-gate -> accNX (straight-line, no branches)
#pragma unroll 2
    for (int kc = 0; kc < IN_ / 32; ++kc) {
      int k0 = kc * 32;
      v16bf a  = load_fragA(ldsA, mrow, k0, LDSE, lane);
      v16bf br = load_fragB(ldsW, jloc,        k0, LDSE, lane);
      v16bf bz = load_fragB(ldsW, 64 + jloc,   k0, LDSE, lane);
      v16bf bn = load_fragB(ldsW, 128 + jloc,  k0, LDSE, lane);
      accR  = wmma_bf16(a, br, accR);
      accZ  = wmma_bf16(a, bz, accZ);
      accNX = wmma_bf16(a, bn, accNX);
    }
    // K in [128,640): hidden part, n-gate -> accNH
#pragma unroll 2
    for (int kc = IN_ / 32; kc < KENC / 32; ++kc) {
      int k0 = kc * 32;
      v16bf a  = load_fragA(ldsA, mrow, k0, LDSE, lane);
      v16bf br = load_fragB(ldsW, jloc,        k0, LDSE, lane);
      v16bf bz = load_fragB(ldsW, 64 + jloc,   k0, LDSE, lane);
      v16bf bn = load_fragB(ldsW, 128 + jloc,  k0, LDSE, lane);
      accR  = wmma_bf16(a, br, accR);
      accZ  = wmma_bf16(a, bz, accZ);
      accNH = wmma_bf16(a, bn, accNH);
    }

    // gates + state update (fp32)
    {
      int j = jglb + (lane & 15);
#pragma unroll
      for (int v = 0; v < 8; ++v) {
        int brow = rowbase + mrow + v + 8 * (lane >> 4);
        float hold = hcur[brow * EH_ + j];
        float r = sigmoidf_(accR[v]);
        float z = sigmoidf_(accZ[v]);
        float n = tanhf(accNX[v] + r * accNH[v]);
        hnxt[brow * EH_ + j] = (1.0f - z) * n + z * hold;
      }
    }
    grid_barrier(bar + 0, bar + 1, NBLK);
  }
}

__global__ void e2g_kernel(const float* __restrict__ hfin, const float* __restrict__ W,
                           float* icp) {
  int gid = blockIdx.x * blockDim.x + threadIdx.x;   // 256*1024 threads
  int b = gid >> 10, n = gid & 1023;
  const float* hp = hfin + b * EH_;
  const float* wp = W + (size_t)n * EH_;
  float acc = 0.0f;
  for (int k = 0; k < EH_; k += 4) {
    float4 h4 = *(const float4*)(hp + k);
    float4 w4 = *(const float4*)(wp + k);
    acc += h4.x * w4.x + h4.y * w4.y + h4.z * w4.z + h4.w * w4.w;
  }
  icp[gid] = acc;
}

__global__ void ic_kernel(const float* __restrict__ icp, const float* __restrict__ eps,
                          float* g0) {
  int gid = blockIdx.x * blockDim.x + threadIdx.x;   // 256*512 threads
  int b = gid >> 9, j = gid & 511;
  float mu = icp[b * 1024 + j];
  float lv = icp[b * 1024 + 512 + j];
  g0[gid] = mu + __expf(0.5f * lv) * eps[gid];
}

__global__ void __launch_bounds__(256)
gen_kernel(const bf16_t* __restrict__ Wru, const bf16_t* __restrict__ Wc,
           float* g0, float* g1, float* rhbuf, float* out, unsigned* bar) {
  __shared__ __align__(16) bf16_t ldsW[192 * LDSG];  // rows: r(64) | u(64) | c(64)
  __shared__ __align__(16) bf16_t ldsA[32 * LDSG];
  const int tid = threadIdx.x;
  const int wgM = blockIdx.x >> 3, wgJ = blockIdx.x & 7;
  const int rowbase = wgM * 32;
  const int w = tid >> 5, lane = tid & 31;
  const int mrow = (w & 1) * 16;
  const int jloc = (w >> 1) * 16;
  const int jglb = wgJ * 64 + jloc;

  // stage WG's weight slice once
  for (int i = tid; i < 192 * (GH_ / 4); i += 256) {
    int rr = i >> 7;                 // / (512/4)
    int k4 = (i & 127) * 4;
    int r64 = rr & 63;
    const bf16_t* gp;
    if (rr < 64)       gp = Wru + (size_t)(wgJ * 64 + r64) * GH_ + k4;        // r rows
    else if (rr < 128) gp = Wru + (size_t)(512 + wgJ * 64 + r64) * GH_ + k4;  // u rows
    else               gp = Wc  + (size_t)(wgJ * 64 + r64) * GH_ + k4;        // c rows
    *(uint2*)(ldsW + rr * LDSG + k4) = *(const uint2*)gp;
  }
  __syncthreads();

#pragma unroll 1
  for (int t = 0; t < T_; ++t) {
    const float* hcur = (t & 1) ? g1 : g0;
    float*       hnxt = (t & 1) ? g0 : g1;

    for (int i = tid; i < 32 * GH_; i += 256) {
      int r = i >> 9, c = i & 511;
      ldsA[r * LDSG + c] = f2bf(hcur[(rowbase + r) * GH_ + c]);
    }
    __syncthreads();

    // GEMM1: ru = h @ Wru^T
    v8f accR = {}, accU = {};
#pragma unroll 2
    for (int kc = 0; kc < GH_ / 32; ++kc) {
      int k0 = kc * 32;
      v16bf a = load_fragA(ldsA, mrow, k0, LDSG, lane);
      accR = wmma_bf16(a, load_fragB(ldsW, jloc,      k0, LDSG, lane), accR);
      accU = wmma_bf16(a, load_fragB(ldsW, 64 + jloc, k0, LDSG, lane), accU);
    }

    float ureg[8], hold[8];
    {
      int j = jglb + (lane & 15);
#pragma unroll
      for (int v = 0; v < 8; ++v) {
        int brow = rowbase + mrow + v + 8 * (lane >> 4);
        float hv = hcur[brow * GH_ + j];
        hold[v] = hv;
        ureg[v] = sigmoidf_(accU[v] + UPDATE_BIAS_);
        rhbuf[brow * GH_ + j] = sigmoidf_(accR[v]) * hv;
      }
    }
    grid_barrier(bar + 2, bar + 3, NBLK);

    // GEMM2: c = tanh((r*h) @ Wc^T)
    for (int i = tid; i < 32 * GH_; i += 256) {
      int r = i >> 9, c = i & 511;
      ldsA[r * LDSG + c] = f2bf(rhbuf[(rowbase + r) * GH_ + c]);
    }
    __syncthreads();

    v8f accC = {};
#pragma unroll 2
    for (int kc = 0; kc < GH_ / 32; ++kc) {
      int k0 = kc * 32;
      v16bf a = load_fragA(ldsA, mrow, k0, LDSG, lane);
      accC = wmma_bf16(a, load_fragB(ldsW, 128 + jloc, k0, LDSG, lane), accC);
    }

    {
      int j = jglb + (lane & 15);
#pragma unroll
      for (int v = 0; v < 8; ++v) {
        int brow = rowbase + mrow + v + 8 * (lane >> 4);
        float c  = tanhf(accC[v]);
        float u  = ureg[v];
        float hn = u * hold[v] + (1.0f - u) * c;
        hnxt[brow * GH_ + j] = hn;
        out[(size_t)brow * (T_ * (size_t)GH_) + (size_t)t * GH_ + j] =
            fminf(fmaxf(hn, -CLIP_VAL_), CLIP_VAL_);
      }
    }
    grid_barrier(bar + 2, bar + 3, NBLK);
  }
}

extern "C" void kernel_launch(void* const* d_in, const int* in_sizes, int n_in,
                              void* d_out, int out_size, void* d_ws, size_t ws_size,
                              hipStream_t stream) {
  const float* src   = (const float*)d_in[0];
  const float* eps   = (const float*)d_in[1];
  const float* encWi = (const float*)d_in[2];
  const float* encWh = (const float*)d_in[3];
  const float* e2gW  = (const float*)d_in[4];
  const float* gWru  = (const float*)d_in[5];
  const float* gWc   = (const float*)d_in[6];
  float* out = (float*)d_out;

  char* ws = (char*)d_ws;
  bf16_t*  Wenc = (bf16_t*)(ws + WS_WENC);
  bf16_t*  Wru  = (bf16_t*)(ws + WS_WRU);
  bf16_t*  Wc   = (bf16_t*)(ws + WS_WC);
  float*   h0   = (float*)(ws + WS_H0);
  float*   h1   = (float*)(ws + WS_H1);
  float*   g0   = (float*)(ws + WS_G0);
  float*   g1   = (float*)(ws + WS_G1);
  float*   rh   = (float*)(ws + WS_RH);
  unsigned* bar = (unsigned*)(ws + WS_BAR);

  float* icp = out + (size_t)B_ * T_ * GH_;  // ic_params after gen_out

  prep_kernel<<<512, 256, 0, stream>>>(encWi, encWh, gWru, gWc, Wenc, Wru, Wc, h0, bar);
  enc_kernel<<<NBLK, 256, 0, stream>>>(src, Wenc, h0, h1, bar);
  e2g_kernel<<<(B_ * 2 * GH_) / 256, 256, 0, stream>>>(h0, e2gW, icp);
  ic_kernel<<<(B_ * GH_) / 256, 256, 0, stream>>>(icp, eps, g0);
  gen_kernel<<<NBLK, 256, 0, stream>>>(Wru, Wc, g0, g1, rh, out, bar);
}
